// CausalAttention_51213190038120
// MI455X (gfx1250) — matrix operand order repaired and checked by reading, not compiled, not run on hardware
//
#include <hip/hip_runtime.h>
#include <hip/hip_bf16.h>
#include <math.h>
#include <stdint.h>

typedef __attribute__((ext_vector_type(16))) __bf16 v16bf;
typedef __attribute__((ext_vector_type(8)))  __bf16 v8bf;
typedef __attribute__((ext_vector_type(4)))  __bf16 v4bf;
typedef __attribute__((ext_vector_type(8)))  float  v8f;
typedef __attribute__((ext_vector_type(4)))  float  v4f;

union Frag16 { v16bf v; v8bf h[2]; };

#define BB 8
#define NNQ 2048
#define DDQ 1024
#define MMQ (BB * NNQ)

static __device__ __forceinline__ v8f wmma_bf16(v16bf a, v16bf b, v8f c) {
  // emits v_wmma_f32_16x16x32_bf16
  return __builtin_amdgcn_wmma_f32_16x16x32_bf16(false, a, false, b, (short)0, c,
                                                 false, false);
}

// ---------------------------------------------------------------------------
// Kernel 1: QKV projections.  C = x(f32)->bf16  @  W(f32)->bf16, f32 accum.
// Block = 256 thr (8 waves), tile 128(M) x 64(N), K staged 32 at a time.
// z=0 -> Q row-major bf16, z=1 -> K row-major bf16, z=2 -> V^T [b][e][n] bf16.
// ---------------------------------------------------------------------------
__global__ __launch_bounds__(256) void proj_qkv(
    const float* __restrict__ x,  const float* __restrict__ Wq,
    const float* __restrict__ Wk, const float* __restrict__ Wv,
    __bf16* __restrict__ qb, __bf16* __restrict__ kb, __bf16* __restrict__ vt)
{
  constexpr int TM = 128, TN = 64, TK = 32, XS = TK + 8;  // padded stride
  __shared__ __align__(16) __bf16 xs [TM * XS];
  __shared__ __align__(16) __bf16 wls[TN * XS];           // W tile, transposed

  const int tid = threadIdx.x;
  const int w   = tid >> 5;
  const int lid = tid & 31;
  const int ln  = lid & 15;
  const int hi  = lid >> 4;
  const int m0  = blockIdx.x * TM;
  const int n0  = blockIdx.y * TN;
  const int z   = blockIdx.z;
  const float* W = (z == 0) ? Wq : (z == 1) ? Wk : Wv;

  v8f acc[4] = {};

  for (int k0 = 0; k0 < DDQ; k0 += TK) {
    // stage x tile 128x32: float4 loads, packed bf16 8B stores
#pragma unroll
    for (int i = 0; i < 4; ++i) {
      int flat = (i * 256 + tid) * 4;
      int r = flat >> 5, c = flat & 31;
      v4f xv = *(const v4f*)&x[(size_t)(m0 + r) * DDQ + k0 + c];
      v4bf xbv;
      xbv.x = (__bf16)xv.x; xbv.y = (__bf16)xv.y;
      xbv.z = (__bf16)xv.z; xbv.w = (__bf16)xv.w;
      *(v4bf*)&xs[r * XS + c] = xbv;
    }
    // stage W tile 32x64 transposed: float4 loads, strided b16 stores
#pragma unroll
    for (int i = 0; i < 2; ++i) {
      int flat = (i * 256 + tid) * 4;
      int kk = flat >> 6, nn = flat & 63;
      v4f wv = *(const v4f*)&W[(size_t)(k0 + kk) * DDQ + n0 + nn];
      wls[(nn + 0) * XS + kk] = (__bf16)wv.x;
      wls[(nn + 1) * XS + kk] = (__bf16)wv.y;
      wls[(nn + 2) * XS + kk] = (__bf16)wv.z;
      wls[(nn + 3) * XS + kk] = (__bf16)wv.w;
    }
    __syncthreads();

    // A fragment: row = w*16 + ln, K contiguous per lane half (k = 16*hi + s)
    Frag16 a;
    const __bf16* ap = &xs[(w * 16 + ln) * XS + 16 * hi];
    a.h[0] = *(const v8bf*)ap;
    a.h[1] = *(const v8bf*)(ap + 8);
#pragma unroll
    for (int g = 0; g < 4; ++g) {
      Frag16 bf;
      const __bf16* bp = &wls[(g * 16 + ln) * XS + 16 * hi];
      bf.h[0] = *(const v8bf*)bp;
      bf.h[1] = *(const v8bf*)(bp + 8);
      acc[g] = wmma_bf16(a.v, bf.v, acc[g]);
    }
    __syncthreads();
  }

  // Store epilogue: destination select hoisted out of the store loops.
  const int mb = m0 + w * 16 + 8 * hi;     // C layout: row = mb + r
  if (z < 2) {
    __bf16* __restrict__ dst = (z == 0) ? qb : kb;
#pragma unroll
    for (int g = 0; g < 4; ++g) {
      int e = n0 + g * 16 + ln;
#pragma unroll
      for (int r = 0; r < 8; ++r)
        dst[(size_t)(mb + r) * DDQ + e] = (__bf16)acc[g][r];
    }
  } else {
    // V^T: the 8 rows of an accumulator are contiguous in n -> one b128 store
    const int bidx  = mb >> 11;            // TM=128 divides 2048: same batch
    const int nbase = mb & (NNQ - 1);
#pragma unroll
    for (int g = 0; g < 4; ++g) {
      int e = n0 + g * 16 + ln;
      v8bf pv;
#pragma unroll
      for (int r = 0; r < 8; ++r) pv[r] = (__bf16)acc[g][r];
      *(v8bf*)&vt[((size_t)bidx * DDQ + e) * NNQ + nbase] = pv;
    }
  }
}

// ---------------------------------------------------------------------------
// Kernel 2: causal flash attention, bf16 WMMA with f32 accumulate.
// Block = 16 query rows, 8 waves. Key mega-tile = 256 (32 per wave).
// Each wave owns 128 output columns; softmax stats merged through LDS.
// ---------------------------------------------------------------------------
__global__ __launch_bounds__(256) void attn_fwd(
    const __bf16* __restrict__ qb, const __bf16* __restrict__ kb,
    const __bf16* __restrict__ vt, float* __restrict__ out)
{
  constexpr int QLD = DDQ + 8;   // 1032, padded bf16 stride
  constexpr int PLD = 256 + 8;   // 264, padded bf16 stride
  __shared__ __align__(16) __bf16 qs[16 * QLD];
  __shared__ __align__(16) __bf16 ps[16 * PLD];
  __shared__ float mred[16][8];
  __shared__ float lred[16][8];

  const int tid = threadIdx.x;
  const int w   = tid >> 5;
  const int lid = tid & 31;
  const int ln  = lid & 15;
  const int hi  = lid >> 4;
  const int b   = blockIdx.y;
  const int q0  = blockIdx.x * 16;

  // Stage 16 query rows (bf16) into LDS with b128 copies
  const __bf16* qg = qb + (size_t)(b * NNQ + q0) * DDQ;
#pragma unroll
  for (int i = 0; i < 8; ++i) {
    int flat = (i * 256 + tid) * 8;
    int r = flat >> 10, c = flat & 1023;
    *(v8bf*)&qs[r * QLD + c] = *(const v8bf*)&qg[(size_t)r * DDQ + c];
  }
  __syncthreads();

  v8f   O[8] = {};
  float mprev[8], lrun[8];
#pragma unroll
  for (int r = 0; r < 8; ++r) { mprev[r] = -__builtin_inff(); lrun[r] = 0.f; }

  const float scale = 0.03125f;                  // 1/sqrt(1024)
  const int   nmt   = (q0 + 16 + 255) >> 8;      // causal: mega-tiles needed
  const __bf16* vg  = vt + (size_t)b * DDQ * NNQ;

  for (int mt = 0; mt < nmt; ++mt) {
    const int kbase = mt * 256 + w * 32;
    const __bf16* kg = kb + (size_t)(b * NNQ + kbase) * DDQ;

    // ---- S = Q K^T for this wave's 32 keys (two 16x16 C frags) ----
    v8f s0 = {}, s1 = {};
#pragma unroll 4
    for (int dk = 0; dk < DDQ; dk += 32) {
      Frag16 a;
      const __bf16* ap = &qs[ln * QLD + dk + 16 * hi];
      a.h[0] = *(const v8bf*)ap;
      a.h[1] = *(const v8bf*)(ap + 8);
      Frag16 b0, b1;
      const __bf16* bp0 = &kg[(size_t)ln * DDQ + dk + 16 * hi];
      b0.h[0] = *(const v8bf*)bp0;
      b0.h[1] = *(const v8bf*)(bp0 + 8);
      s0 = wmma_bf16(a.v, b0.v, s0);
      const __bf16* bp1 = &kg[(size_t)(ln + 16) * DDQ + dk + 16 * hi];
      b1.h[0] = *(const v8bf*)bp1;
      b1.h[1] = *(const v8bf*)(bp1 + 8);
      s1 = wmma_bf16(a.v, b1.v, s1);
    }

    // ---- scale + causal mask + per-chunk row max ----
    float mc[8];
#pragma unroll
    for (int r = 0; r < 8; ++r) {
      int qrow = q0 + r + 8 * hi;
      int k0i  = kbase + ln;
      float v0 = (k0i      <= qrow) ? s0[r] * scale : -__builtin_inff();
      float v1 = (k0i + 16 <= qrow) ? s1[r] * scale : -__builtin_inff();
      s0[r] = v0; s1[r] = v1;
      mc[r] = fmaxf(v0, v1);
    }
#pragma unroll
    for (int off = 1; off < 16; off <<= 1)
#pragma unroll
      for (int r = 0; r < 8; ++r)
        mc[r] = fmaxf(mc[r], __shfl_xor(mc[r], off, 32));
    if (ln == 0)
#pragma unroll
      for (int r = 0; r < 8; ++r) mred[r + 8 * hi][w] = mc[r];
    __syncthreads();

    // ---- combined new max; P = exp(s - m_new) written bf16 to LDS ----
    float mnew[8], lc[8];
#pragma unroll
    for (int r = 0; r < 8; ++r) {
      float m = mprev[r];
#pragma unroll
      for (int w2 = 0; w2 < 8; ++w2) m = fmaxf(m, mred[r + 8 * hi][w2]);
      mnew[r] = m;
      float p0 = __expf(s0[r] - m);
      float p1 = __expf(s1[r] - m);
      ps[(r + 8 * hi) * PLD + w * 32 + ln]      = (__bf16)p0;
      ps[(r + 8 * hi) * PLD + w * 32 + ln + 16] = (__bf16)p1;
      lc[r] = p0 + p1;
    }
#pragma unroll
    for (int off = 1; off < 16; off <<= 1)
#pragma unroll
      for (int r = 0; r < 8; ++r) lc[r] += __shfl_xor(lc[r], off, 32);
    if (ln == 0)
#pragma unroll
      for (int r = 0; r < 8; ++r) lred[r + 8 * hi][w] = lc[r];
    __syncthreads();

    // ---- rescale running O / l ----
#pragma unroll
    for (int r = 0; r < 8; ++r) {
      float f  = __expf(mprev[r] - mnew[r]);
      float ls = 0.f;
#pragma unroll
      for (int w2 = 0; w2 < 8; ++w2) ls += lred[r + 8 * hi][w2];
      lrun[r]  = lrun[r] * f + ls;
      mprev[r] = mnew[r];
#pragma unroll
      for (int g = 0; g < 8; ++g) O[g][r] *= f;
    }

    // ---- O += P @ V  (256-key depth, wave's private 128 output cols) ----
#pragma unroll 2
    for (int kc = 0; kc < 8; ++kc) {
      Frag16 a;
      const __bf16* ap = &ps[ln * PLD + kc * 32 + 16 * hi];
      a.h[0] = *(const v8bf*)ap;
      a.h[1] = *(const v8bf*)(ap + 8);
      const int nb = mt * 256 + kc * 32 + 16 * hi;
#pragma unroll
      for (int g = 0; g < 8; ++g) {
        int e = w * 128 + g * 16 + ln;
        Frag16 bbv;
        const __bf16* bp = &vg[(size_t)e * NNQ + nb];
        bbv.h[0] = *(const v8bf*)bp;
        bbv.h[1] = *(const v8bf*)(bp + 8);
        O[g] = wmma_bf16(a.v, bbv.v, O[g]);
      }
    }
    __syncthreads();
  }

  // ---- epilogue: divide by row sum, non-temporal f32 stores ----
  float inv[8];
#pragma unroll
  for (int r = 0; r < 8; ++r) inv[r] = 1.0f / lrun[r];
#pragma unroll
  for (int g = 0; g < 8; ++g) {
    int e = w * 128 + g * 16 + ln;
#pragma unroll
    for (int r = 0; r < 8; ++r) {
      float* p = &out[(size_t)(b * NNQ + q0 + r + 8 * hi) * DDQ + e];
      __builtin_nontemporal_store(O[g][r] * inv[r], p);
    }
  }
}

// ---------------------------------------------------------------------------
extern "C" void kernel_launch(void* const* d_in, const int* in_sizes, int n_in,
                              void* d_out, int out_size, void* d_ws, size_t ws_size,
                              hipStream_t stream) {
  const float* x  = (const float*)d_in[0];
  const float* Wq = (const float*)d_in[1];
  const float* Wk = (const float*)d_in[2];
  const float* Wv = (const float*)d_in[3];
  float* out = (float*)d_out;

  char* ws = (char*)d_ws;
  const size_t sz = (size_t)MMQ * DDQ * sizeof(__bf16);  // 33.5 MB each
  __bf16* qb = (__bf16*)(ws);
  __bf16* kb = (__bf16*)(ws + sz);
  __bf16* vt = (__bf16*)(ws + 2 * sz);

  dim3 pgrid(MMQ / 128, DDQ / 64, 3);
  proj_qkv<<<pgrid, 256, 0, stream>>>(x, Wq, Wk, Wv, qb, kb, vt);

  dim3 agrid(NNQ / 16, BB);
  attn_fwd<<<agrid, 256, 0, stream>>>(qb, kb, vt, out);
}
